// GCN_10694468567401
// MI455X (gfx1250) — matrix operand order repaired and checked by reading, not compiled, not run on hardware
//
#include <hip/hip_runtime.h>

#define N_NODES 50000
#define N_EDGES 800000
#define ROW_TILES 3125   // N_NODES / 16 exactly

typedef __attribute__((ext_vector_type(16))) __bf16 v16bf;
typedef __attribute__((ext_vector_type(8)))  float  v8f;

// ---------------------------------------------------------------- utilities
__global__ void zero_kernel(float* __restrict__ p, int n) {
  int i = blockIdx.x * blockDim.x + threadIdx.x;
  if (i < n) p[i] = 0.0f;
}

__global__ void degree_kernel(const int* __restrict__ snd, const int* __restrict__ rcv,
                              float* __restrict__ sdeg, float* __restrict__ rdeg) {
  int e = blockIdx.x * blockDim.x + threadIdx.x;
  if (e >= N_EDGES) return;
  atomicAdd(&sdeg[snd[e]], 1.0f);
  atomicAdd(&rdeg[rcv[e]], 1.0f);
}

__global__ void norm_kernel(const float* __restrict__ sdeg, const float* __restrict__ rdeg,
                            float* __restrict__ snorm, float* __restrict__ rnorm) {
  int i = blockIdx.x * blockDim.x + threadIdx.x;
  if (i >= N_NODES) return;
  snorm[i] = rsqrtf(fmaxf(sdeg[i], 1.0f));
  rnorm[i] = rsqrtf(fmaxf(rdeg[i], 1.0f));
}

// ------------------------------------------------- WMMA fragment packing
// A (16x32 bf16) per-lane layout (ISA 7.12.2): row M = lane&15, half = lane>>4,
// element i -> VGPR p=i>>1, K = (p<4 ? 2p : 2p+8) + 8*half + (i&1).
__device__ __forceinline__ int a_koff(int i, int half) {
  int p = i >> 1;
  return (p < 4 ? 2 * p : 2 * p + 8) + 8 * half + (i & 1);
}

// Pack f32 activations [N x Fin] row-major -> bf16 A fragments.
__global__ void pack_x_kernel(const float* __restrict__ x, __bf16* __restrict__ apack, int Fin) {
  int ks = Fin >> 5;
  int idx = blockIdx.x * blockDim.x + threadIdx.x;   // ((r*ks + s)*32 + lane)
  if (idx >= ROW_TILES * ks * 32) return;
  int lane = idx & 31;
  int s    = (idx >> 5) % ks;
  int r    = idx / (32 * ks);
  int m = lane & 15, half = lane >> 4;
  const float* xr = x + (size_t)(r * 16 + m) * Fin + s * 32;
  v16bf v;
#pragma unroll
  for (int i = 0; i < 16; ++i) v[i] = (__bf16)xr[a_koff(i, half)];
  *(v16bf*)(apack + (size_t)idx * 16) = v;
}

// Fused: x_next = relu(agg * rnorm) packed straight into bf16 A fragments.
__global__ void pack_relu_kernel(const float* __restrict__ agg, const float* __restrict__ rnorm,
                                 __bf16* __restrict__ apack, int F) {
  int ks = F >> 5;
  int idx = blockIdx.x * blockDim.x + threadIdx.x;
  if (idx >= ROW_TILES * ks * 32) return;
  int lane = idx & 31;
  int s    = (idx >> 5) % ks;
  int r    = idx / (32 * ks);
  int m = lane & 15, half = lane >> 4;
  int row = r * 16 + m;
  float rn = rnorm[row];
  const float* ar = agg + (size_t)row * F + s * 32;
  v16bf v;
#pragma unroll
  for (int i = 0; i < 16; ++i) v[i] = (__bf16)fmaxf(ar[a_koff(i, half)] * rn, 0.0f);
  *(v16bf*)(apack + (size_t)idx * 16) = v;
}

// B (32x16 bf16) per-lane layout: col N = lane&15, half = lane>>4,
// element i -> K = a_koff(i, half) (same K order as A).  W is [Fin x Fout] row-major.
__global__ void pack_w_kernel(const float* __restrict__ W, __bf16* __restrict__ wp,
                              int Fin, int Fout) {
  int ks = Fin >> 5, nt = Fout >> 4;
  int idx = blockIdx.x * blockDim.x + threadIdx.x;   // ((t*ks + s)*32 + lane)
  if (idx >= nt * ks * 32) return;
  int lane = idx & 31;
  int s    = (idx >> 5) % ks;
  int t    = idx / (32 * ks);
  int n = t * 16 + (lane & 15);
  int half = lane >> 4;
  v16bf v;
#pragma unroll
  for (int i = 0; i < 16; ++i) {
    int k = s * 32 + a_koff(i, half);
    v[i] = (__bf16)W[(size_t)k * Fout + n];
  }
  *(v16bf*)(wp + (size_t)idx * 16) = v;
}

// ------------------------------------------------- GEMM: h = (x@W + b) * snorm
// 256 threads = 8 waves; waves tiled 2 (rows) x 4 (cols); each wave owns a
// 16x64 output strip = 4 WMMA accumulator tiles, reusing the A fragment.
__global__ void gemm_wmma_kernel(const __bf16* __restrict__ apack,
                                 const __bf16* __restrict__ wpack,
                                 const float* __restrict__ bias,
                                 const float* __restrict__ snorm,
                                 float* __restrict__ h, int ks, int Fout) {
  int lane = threadIdx.x & 31;
  int w    = threadIdx.x >> 5;
  int wr = w >> 2, wc = w & 3;
  int r = blockIdx.x * 2 + wr;
  if (r >= ROW_TILES) return;            // uniform per wave -> EXEC stays all-1s
  int ntiles = Fout >> 4;
  int t0 = wc * 4;
  if (t0 >= ntiles) return;              // uniform per wave

  const v16bf* A = (const v16bf*)apack + (size_t)r * ks * 32 + lane;
  const v16bf* B = (const v16bf*)wpack + lane;

  v8f acc0 = {}, acc1 = {}, acc2 = {}, acc3 = {};
  for (int s = 0; s < ks; ++s) {
    v16bf a = A[(size_t)s * 32];
    if (s + 1 < ks) __builtin_prefetch(&A[(size_t)(s + 1) * 32], 0, 1);
    v16bf b0 = B[(size_t)((t0 + 0) * ks + s) * 32];
    v16bf b1 = B[(size_t)((t0 + 1) * ks + s) * 32];
    v16bf b2 = B[(size_t)((t0 + 2) * ks + s) * 32];
    v16bf b3 = B[(size_t)((t0 + 3) * ks + s) * 32];
    acc0 = __builtin_amdgcn_wmma_f32_16x16x32_bf16(false, a, false, b0, (short)0, acc0, false, false);
    acc1 = __builtin_amdgcn_wmma_f32_16x16x32_bf16(false, a, false, b1, (short)0, acc1, false, false);
    acc2 = __builtin_amdgcn_wmma_f32_16x16x32_bf16(false, a, false, b2, (short)0, acc2, false, false);
    acc3 = __builtin_amdgcn_wmma_f32_16x16x32_bf16(false, a, false, b3, (short)0, acc3, false, false);
  }

  // C/D layout: col N = lane&15, VGPR v -> row M = v + 8*(lane>>4).
  int half = lane >> 4, nl = lane & 15;
  int c0 = (t0 + 0) * 16 + nl, c1 = (t0 + 1) * 16 + nl;
  int c2 = (t0 + 2) * 16 + nl, c3 = (t0 + 3) * 16 + nl;
  float bb0 = bias[c0], bb1 = bias[c1], bb2 = bias[c2], bb3 = bias[c3];
#pragma unroll
  for (int v = 0; v < 8; ++v) {
    int row = r * 16 + v + 8 * half;
    float sn = snorm[row];
    float* hr = h + (size_t)row * Fout;
    hr[c0] = (acc0[v] + bb0) * sn;
    hr[c1] = (acc1[v] + bb1) * sn;
    hr[c2] = (acc2[v] + bb2) * sn;
    hr[c3] = (acc3[v] + bb3) * sn;
  }
}

// ------------------------------------------------- edge scatter-add
__global__ void scatter_kernel(const float* __restrict__ h, const int* __restrict__ snd,
                               const int* __restrict__ rcv, float* __restrict__ agg, int F) {
  int chunks = F >> 2;
  int idx = blockIdx.x * blockDim.x + threadIdx.x;
  if (idx >= N_EDGES * chunks) return;
  int e = idx / chunks;
  int c = (idx - e * chunks) * 4;
  int s = snd[e], r = rcv[e];
  float4 v = *(const float4*)(h + (size_t)s * F + c);
  float* dst = agg + (size_t)r * F + c;
  atomicAdd(dst + 0, v.x);
  atomicAdd(dst + 1, v.y);
  atomicAdd(dst + 2, v.z);
  atomicAdd(dst + 3, v.w);
}

// Final layer epilogue: out = relu(agg * rnorm), f32.
__global__ void final_kernel(const float* __restrict__ agg, const float* __restrict__ rnorm,
                             float* __restrict__ out, int F) {
  int i = blockIdx.x * blockDim.x + threadIdx.x;
  if (i >= N_NODES * F) return;
  int row = i / F;
  out[i] = fmaxf(agg[i] * rnorm[row], 0.0f);
}

// ---------------------------------------------------------------- launch
extern "C" void kernel_launch(void* const* d_in, const int* in_sizes, int n_in,
                              void* d_out, int out_size, void* d_ws, size_t ws_size,
                              hipStream_t stream) {
  const float* nodes   = (const float*)d_in[0];
  const int*   senders = (const int*)d_in[1];
  const int*   recvs   = (const int*)d_in[2];
  const float* W1 = (const float*)d_in[3]; const float* b1 = (const float*)d_in[4];
  const float* W2 = (const float*)d_in[5]; const float* b2 = (const float*)d_in[6];
  const float* W3 = (const float*)d_in[7]; const float* b3 = (const float*)d_in[8];

  char* ws = (char*)d_ws;
  size_t off = 0;
  auto alloc = [&](size_t bytes) { void* p = ws + off; off = (off + bytes + 255) & ~(size_t)255; return p; };
  float* sdeg  = (float*)alloc((size_t)N_NODES * 4);
  float* rdeg  = (float*)alloc((size_t)N_NODES * 4);
  float* snorm = (float*)alloc((size_t)N_NODES * 4);
  float* rnorm = (float*)alloc((size_t)N_NODES * 4);
  __bf16* wp1  = (__bf16*)alloc((size_t)16 * 4 * 32 * 16 * 2);   //  64 KB
  __bf16* wp2  = (__bf16*)alloc((size_t)16 * 8 * 32 * 16 * 2);   // 128 KB
  __bf16* wp3  = (__bf16*)alloc((size_t)8  * 8 * 32 * 16 * 2);   //  64 KB
  __bf16* apack = (__bf16*)alloc((size_t)ROW_TILES * 8 * 32 * 16 * 2); // 25.6 MB
  float* h     = (float*)alloc((size_t)N_NODES * 256 * 4);       // 51.2 MB
  float* agg   = (float*)alloc((size_t)N_NODES * 256 * 4);       // 51.2 MB

  const int T = 256;
  auto blocks = [](long long n, int t) { return (int)((n + t - 1) / t); };

  // degrees + norms
  zero_kernel<<<blocks(2LL * N_NODES, T), T, 0, stream>>>(sdeg, 2 * N_NODES);
  degree_kernel<<<blocks(N_EDGES, T), T, 0, stream>>>(senders, recvs, sdeg, rdeg);
  norm_kernel<<<blocks(N_NODES, T), T, 0, stream>>>(sdeg, rdeg, snorm, rnorm);

  // weight packing (bf16 WMMA B-fragment layout)
  pack_w_kernel<<<blocks(16LL * 4 * 32, T), T, 0, stream>>>(W1, wp1, 128, 256);
  pack_w_kernel<<<blocks(16LL * 8 * 32, T), T, 0, stream>>>(W2, wp2, 256, 256);
  pack_w_kernel<<<blocks(8LL * 8 * 32, T), T, 0, stream>>>(W3, wp3, 256, 128);

  // input activations -> A fragments
  pack_x_kernel<<<blocks((long long)ROW_TILES * 4 * 32, T), T, 0, stream>>>(nodes, apack, 128);

  const int gemm_grid = (ROW_TILES + 1) / 2;

  // ---- layer 1: 128 -> 256
  zero_kernel<<<blocks((long long)N_NODES * 256, T), T, 0, stream>>>(agg, N_NODES * 256);
  gemm_wmma_kernel<<<gemm_grid, 256, 0, stream>>>(apack, wp1, b1, snorm, h, 4, 256);
  scatter_kernel<<<blocks((long long)N_EDGES * 64, T), T, 0, stream>>>(h, senders, recvs, agg, 256);
  pack_relu_kernel<<<blocks((long long)ROW_TILES * 8 * 32, T), T, 0, stream>>>(agg, rnorm, apack, 256);

  // ---- layer 2: 256 -> 256
  zero_kernel<<<blocks((long long)N_NODES * 256, T), T, 0, stream>>>(agg, N_NODES * 256);
  gemm_wmma_kernel<<<gemm_grid, 256, 0, stream>>>(apack, wp2, b2, snorm, h, 8, 256);
  scatter_kernel<<<blocks((long long)N_EDGES * 64, T), T, 0, stream>>>(h, senders, recvs, agg, 256);
  pack_relu_kernel<<<blocks((long long)ROW_TILES * 8 * 32, T), T, 0, stream>>>(agg, rnorm, apack, 256);

  // ---- layer 3: 256 -> 128
  zero_kernel<<<blocks((long long)N_NODES * 128, T), T, 0, stream>>>(agg, N_NODES * 128);
  gemm_wmma_kernel<<<gemm_grid, 256, 0, stream>>>(apack, wp3, b3, snorm, h, 8, 128);
  scatter_kernel<<<blocks((long long)N_EDGES * 32, T), T, 0, stream>>>(h, senders, recvs, agg, 128);
  final_kernel<<<blocks((long long)N_NODES * 128, T), T, 0, stream>>>(agg, rnorm, (float*)d_out, 128);

  (void)in_sizes; (void)n_in; (void)out_size; (void)ws_size;
}
